// TCNAttentionBlock_13477607375366
// MI455X (gfx1250) — matrix operand order repaired and checked by reading, not compiled, not run on hardware
//
#include <hip/hip_runtime.h>

// ---------------------------------------------------------------------------
// Problem constants (from the reference)
// ---------------------------------------------------------------------------
namespace {
constexpr int kB = 8;
constexpr int kS = 1024;
constexpr int kC = 512;
constexpr int kH = 8;
constexpr int kD = 64;          // kC / kH
constexpr int kP = 16;          // persistent tokens
constexpr int kK = 3;           // conv kernel
constexpr int kSPAD = 1056;     // (S + P) padded to a multiple of 32
constexpr int kLTiles = kSPAD / 32;   // 33 key tiles of 32
}

typedef _Float16 half_t;
typedef __attribute__((ext_vector_type(16))) _Float16 v16h;
typedef __attribute__((ext_vector_type(8)))  _Float16 v8h;
typedef __attribute__((ext_vector_type(8)))  float    v8f;

// ---------------------------------------------------------------------------
// WMMA wrapper: D = A(16x32 f16) * B(32x16 f16) + C(16x16 f32)
// ---------------------------------------------------------------------------
__device__ __forceinline__ v8f wmma_ff(v16h a, v16h b, v8f c) {
  return __builtin_amdgcn_wmma_f32_16x16x32_f16(
      /*neg_a=*/false, a, /*neg_b=*/false, b,
      /*c_mod=*/(short)0, c, /*reuse_a=*/false, /*reuse_b=*/false);
}

// ---------------------------------------------------------------------------
// Tensor Data Mover: 1-D contiguous tile (N halfwords) global -> LDS.
// D# per CDNA5 ISA ch.8: group0 = {count/type, lds_addr, global_addr},
// group1 = {data_size, tensor_dim0/1, tile_dim0, stride}.
// This toolchain's builtin takes 6 args:
//   (uint32x4 g0, int32x8 g1, int32x4, int32x4, int32x8, i32 cpol)
// ---------------------------------------------------------------------------
__device__ __forceinline__ void tdm_load_1d(const void* gsrc, unsigned lds_off,
                                            unsigned n_half) {
#if __has_builtin(__builtin_amdgcn_tensor_load_to_lds)
  typedef __attribute__((ext_vector_type(4))) unsigned int u32x4;
  typedef __attribute__((ext_vector_type(8))) int i32x8;
  typedef __attribute__((ext_vector_type(4))) int i32x4;
  unsigned long long ga = (unsigned long long)(uintptr_t)gsrc;
  u32x4 g0;
  g0[0] = 1u;                                          // count=1 (user D#)
  g0[1] = lds_off;                                     // lds_addr (bytes)
  g0[2] = (unsigned)(ga & 0xffffffffu);                // global_addr[31:0]
  g0[3] = (unsigned)((ga >> 32) & 0x01ffffffu)         // global_addr[56:32]
          | (2u << 30);                                // type = 2 ("image")
  i32x8 g1;
  g1[0] = (1 << 16);                                   // data_size=1 (2 bytes)
  g1[1] = (int)((n_half & 0xffffu) << 16);             // tensor_dim0 lo16
  g1[2] = (int)((n_half >> 16) & 0xffffu)              // tensor_dim0 hi16
          | (1 << 16);                                 // tensor_dim1 = 1
  g1[3] = (int)((n_half & 0xffffu) << 16);             // tile_dim0 = N
  g1[4] = 0;                                           // tile_dim1/2 unused
  g1[5] = (int)n_half;                                 // tensor_dim0_stride
  g1[6] = 0;
  g1[7] = 0;
  i32x4 z4 = {0, 0, 0, 0};
  i32x8 z8 = {0, 0, 0, 0, 0, 0, 0, 0};
  __builtin_amdgcn_tensor_load_to_lds(g0, g1, z4, z4, z8, 0);
#else
  (void)gsrc; (void)lds_off; (void)n_half;
#endif
}

// Stage n_half contiguous f16 elements from global into LDS (whole block
// participates; caller must __syncthreads() afterwards).
__device__ __forceinline__ void stage_to_lds(half_t* lds_dst,
                                             const half_t* __restrict__ gsrc,
                                             unsigned n_half) {
#if __has_builtin(__builtin_amdgcn_tensor_load_to_lds)
  if (threadIdx.x < 32) {    // wave 0 issues the DMA (TDM ignores EXEC)
    // LDS-aperture flat address carries the LDS byte offset in addr[31:0].
    tdm_load_1d(gsrc, (unsigned)(uintptr_t)lds_dst, n_half);
#if __has_builtin(__builtin_amdgcn_s_wait_tensorcnt)
    __builtin_amdgcn_s_wait_tensorcnt(0);
#endif
  }
#else
  const v8h* s = (const v8h*)gsrc;
  v8h* d = (v8h*)lds_dst;
  for (unsigned i = threadIdx.x; i < n_half / 8; i += blockDim.x) d[i] = s[i];
#endif
}

// ---------------------------------------------------------------------------
// Operand loaders following the CDNA5 ISA VGPR layouts (wave32).
// A 16x32 f16 (MxK): lanes 0-15 row=lane, K {0..7,16..23};
//                    lanes 16-31 row=lane-16, K {8..15,24..31}.
// ---------------------------------------------------------------------------
__device__ __forceinline__ v16h load_A_f16(const half_t* __restrict__ src,
                                           int ld, int row0, int col0) {
  int lane = threadIdx.x & 31;
  int cn = lane & 15, hf = lane >> 4;
  const half_t* p = src + (size_t)(row0 + cn) * ld + col0 + hf * 8;
  v8h lo = *(const v8h*)(p);
  v8h hi = *(const v8h*)(p + 16);
  v16h a;
#pragma unroll
  for (int i = 0; i < 8; ++i) { a[i] = lo[i]; a[i + 8] = hi[i]; }
  return a;
}

// Same A layout, source is f32 (converted on the fly).
__device__ __forceinline__ v16h load_A_f32src(const float* __restrict__ src,
                                              int ld, int row0, int col0) {
  int lane = threadIdx.x & 31;
  int cn = lane & 15, hf = lane >> 4;
  const float* p = src + (size_t)(row0 + cn) * ld + col0 + hf * 8;
  v8f lo = *(const v8f*)(p);
  v8f hi = *(const v8f*)(p + 16);
  v16h a;
#pragma unroll
  for (int i = 0; i < 8; ++i) { a[i] = (half_t)lo[i]; a[i + 8] = (half_t)hi[i]; }
  return a;
}

// B 32x16 f16 (KxN): column `cn` lives in lane cn (K {0..7,16..23}) and lane
// cn+16 (K {8..15,24..31}); base points at element (k0, col0); columns are contiguous with
// stride col_stride along K.
__device__ __forceinline__ v16h load_B16(const half_t* __restrict__ base,
                                         size_t col_stride) {
  int lane = threadIdx.x & 31;
  int cn = lane & 15, hf = lane >> 4;
  const half_t* p = base + (size_t)cn * col_stride + hf * 8;
  v8h lo = *(const v8h*)(p);
  v8h hi = *(const v8h*)(p + 16);
  v16h b;
#pragma unroll
  for (int i = 0; i < 8; ++i) { b[i] = lo[i]; b[i + 8] = hi[i]; }
  return b;
}

// C/D 16x16 f32: VGPR i, lanes 0-15 -> (row i, col lane); lanes 16-31 ->
// (row i+8, col lane-16).
__device__ __forceinline__ void store_C_f16(half_t* __restrict__ dst, int ld,
                                            int row0, int col0, v8f c) {
  int lane = threadIdx.x & 31;
  int cn = lane & 15, hf = lane >> 4;
#pragma unroll
  for (int i = 0; i < 8; ++i)
    dst[(size_t)(row0 + hf * 8 + i) * ld + col0 + cn] = (half_t)c[i];
}

// Transposed f16 store: element (row,col) -> dst[col * ldT + row].
__device__ __forceinline__ void store_CT_f16(half_t* __restrict__ dst, int ldT,
                                             int row0, int col0, v8f c) {
  int lane = threadIdx.x & 31;
  int cn = lane & 15, hf = lane >> 4;
#pragma unroll
  for (int i = 0; i < 8; ++i)
    dst[(size_t)(col0 + cn) * ldT + row0 + hf * 8 + i] = (half_t)c[i];
}

// Row-wise reductions inside each 16-lane half of a wave32.
__device__ __forceinline__ float rowmax16(float v) {
#pragma unroll
  for (int m = 1; m < 16; m <<= 1) v = fmaxf(v, __shfl_xor(v, m, 32));
  return v;
}
__device__ __forceinline__ float rowsum16(float v) {
#pragma unroll
  for (int m = 1; m < 16; m <<= 1) v += __shfl_xor(v, m, 32);
  return v;
}

// ---------------------------------------------------------------------------
// K0a: generic f32 -> f16 converter (for Wq/Wk/Wv/W_fc)
// ---------------------------------------------------------------------------
__global__ void convert_f16_kernel(const float* __restrict__ src,
                                   half_t* __restrict__ dst, int n) {
  int i = blockIdx.x * blockDim.x + threadIdx.x;
  if (i < n) dst[i] = (half_t)src[i];
}

// K0b: repack conv weight (C,C,K) f32 -> (K,C,C) f16 as wB[t][co][ci]
__global__ void pack_conv_kernel(const float* __restrict__ w,
                                 half_t* __restrict__ dst) {
  int i = blockIdx.x * blockDim.x + threadIdx.x;
  if (i >= kK * kC * kC) return;
  int t = i / (kC * kC);
  int rem = i - t * kC * kC;
  int co = rem / kC;
  int ci = rem - co * kC;
  dst[i] = (half_t)w[((size_t)co * kC + ci) * kK + t];
}

// ---------------------------------------------------------------------------
// K0c: append persistent K/V (broadcast over batch & heads), zero padding tail
// ---------------------------------------------------------------------------
__global__ void persist_kv_kernel(const float* __restrict__ p_keys,
                                  const float* __restrict__ p_values,
                                  half_t* __restrict__ k16,
                                  half_t* __restrict__ vT16) {
  int idx = blockIdx.x * blockDim.x + threadIdx.x;   // 0 .. 262143
  int seg = idx >> 16;
  int t = idx & 65535;
  if (seg == 0) {                       // k16[b, S+p, h, d] = p_keys[p,0,d]
    int d = t & 63, h = (t >> 6) & 7, p = (t >> 9) & 15, b = t >> 13;
    k16[((size_t)b * kSPAD + kS + p) * kC + h * kD + d] =
        (half_t)p_keys[p * kD + d];
  } else if (seg == 1) {                // vT16[b, h, d, S+p] = p_values[p,0,d]
    int d = t & 63, h = (t >> 6) & 7, p = (t >> 9) & 15, b = t >> 13;
    vT16[(((size_t)b * kH + h) * kD + d) * kSPAD + kS + p] =
        (half_t)p_values[p * kD + d];
  } else if (seg == 2) {                // zero k16 pad rows [S+P, SPAD)
    int c = t & 511, row = (t >> 9) & 15, b = t >> 13;
    k16[((size_t)b * kSPAD + kS + kP + row) * kC + c] = (half_t)0.f;
  } else {                              // zero vT16 pad cols [S+P, SPAD)
    int l = t & 15, d = (t >> 4) & 63, h = (t >> 10) & 7, b = t >> 13;
    vT16[(((size_t)b * kH + h) * kD + d) * kSPAD + kS + kP + l] = (half_t)0.f;
  }
}

// ---------------------------------------------------------------------------
// K1: QKV projections.  One wave per (b, h, 16-row tile).
//   q16 [b,s,h,d], k16 [b,l,h,d] (rows 0..S-1), vT16 [b,h,d,l]
// ---------------------------------------------------------------------------
__global__ __launch_bounds__(256) void qkv_kernel(
    const float* __restrict__ x, const half_t* __restrict__ wq16,
    const half_t* __restrict__ wk16, const half_t* __restrict__ wv16,
    half_t* __restrict__ q16, half_t* __restrict__ k16,
    half_t* __restrict__ vT16) {
  int wave = threadIdx.x >> 5;
  int task = blockIdx.x * (blockDim.x >> 5) + wave;   // 4096 tasks
  int qt = task & 63;
  int h = (task >> 6) & 7;
  int b = task >> 9;
  int s0 = qt * 16;

  const float* xh = x + (size_t)b * kS * kC + h * kD;   // ld = kC
  v16h A0 = load_A_f32src(xh, kC, s0, 0);
  v16h A1 = load_A_f32src(xh, kC, s0, 32);

  const half_t* Ws[3] = {wq16, wk16, wv16};
#pragma unroll
  for (int proj = 0; proj < 3; ++proj) {
    const half_t* W = Ws[proj];
#pragma unroll
    for (int n = 0; n < 4; ++n) {
      v8f acc = {};
      acc = wmma_ff(A0, load_B16(W + (size_t)(n * 16) * kD + 0, kD), acc);
      acc = wmma_ff(A1, load_B16(W + (size_t)(n * 16) * kD + 32, kD), acc);
      if (proj == 0) {
        store_C_f16(q16 + (size_t)b * kS * kC, kC, s0, h * kD + n * 16, acc);
      } else if (proj == 1) {
        store_C_f16(k16 + (size_t)b * kSPAD * kC, kC, s0, h * kD + n * 16, acc);
      } else {
        store_CT_f16(vT16 + ((size_t)b * kH + h) * kD * kSPAD, kSPAD,
                     s0, n * 16, acc);
      }
    }
  }
}

// ---------------------------------------------------------------------------
// K2: fused head-mixed flash attention.
//   8 waves / block = 8 heads; one block per (b, q-tile of 16).
// ---------------------------------------------------------------------------
__device__ __forceinline__ void compute_smix(
    const v16h& QA0, const v16h& QA1, const half_t* __restrict__ kbase,
    const float* wpre, float wslope, const unsigned char* __restrict__ pm,
    float (*sh_e)[16][32], int j, int s0, int l0, float smix[2][8]) {
  const float inv_sqrt_c = 0.044194173824159216f;   // 1/sqrt(512)
  int lane = threadIdx.x & 31;
  int cn = lane & 15, hf = lane >> 4;

  v8f e0 = {}, e1 = {};
  e0 = wmma_ff(QA0, load_B16(kbase + (size_t)(l0 + 0) * kC + 0, kC), e0);
  e0 = wmma_ff(QA1, load_B16(kbase + (size_t)(l0 + 0) * kC + 32, kC), e0);
  e1 = wmma_ff(QA0, load_B16(kbase + (size_t)(l0 + 16) * kC + 0, kC), e1);
  e1 = wmma_ff(QA1, load_B16(kbase + (size_t)(l0 + 16) * kC + 32, kC), e1);

  __syncthreads();            // prior readers of sh_e are done
#pragma unroll
  for (int i = 0; i < 8; ++i) {
    sh_e[j][hf * 8 + i][cn] = e0[i];
    sh_e[j][hf * 8 + i][16 + cn] = e1[i];
  }
  __syncthreads();

#pragma unroll
  for (int c = 0; c < 2; ++c) {
#pragma unroll
    for (int i = 0; i < 8; ++i) {
      float v = 0.f;
#pragma unroll
      for (int hh = 0; hh < 8; ++hh)
        v = fmaf(wpre[hh], sh_e[hh][hf * 8 + i][c * 16 + cn], v);
      int lg = l0 + c * 16 + cn;
      int qg = s0 + hf * 8 + i;
      if (lg < kS) v -= fabsf((float)(qg - lg)) * wslope;   // mixed ALiBi
      bool valid = (lg < kS) ? (pm[lg] == 0) : (lg < kS + kP);
      if (!valid) v = -10000.0f;
      smix[c][i] = v * inv_sqrt_c;
    }
  }
}

__global__ __launch_bounds__(256) void attn_kernel(
    const half_t* __restrict__ q16, const half_t* __restrict__ k16,
    const half_t* __restrict__ vT16, const float* __restrict__ W_pre,
    const float* __restrict__ W_post, const unsigned char* __restrict__ pad_mask,
    half_t* __restrict__ out16) {
  __shared__ __align__(16) float  sh_e[8][16][32];   // raw energies / probs
  __shared__ __align__(16) half_t sh_a[8][16][32];   // mixed attn (A operand)

  int wave = threadIdx.x >> 5;
  int lane = threadIdx.x & 31;
  int cn = lane & 15, hf = lane >> 4;
  int qt = blockIdx.x & 63;
  int b = blockIdx.x >> 6;
  int s0 = qt * 16;
  int j = wave;                                      // this wave's head

  float wpre[8], wpost[8];
#pragma unroll
  for (int i = 0; i < 8; ++i) {
    wpre[i] = W_pre[j * 8 + i];
    wpost[i] = W_post[j * 8 + i];
  }
  float wslope = 0.f;
#pragma unroll
  for (int i = 0; i < 8; ++i) wslope += wpre[i] * exp2f(-(float)(i + 1));

  const unsigned char* pm = pad_mask + (size_t)b * kS;
  const half_t* qbase = q16 + (size_t)b * kS * kC + j * kD;
  const half_t* kbase = k16 + (size_t)b * kSPAD * kC + j * kD;
  const half_t* vbase = vT16 + ((size_t)b * kH + j) * kD * kSPAD;

  v16h QA0 = load_A_f16(qbase, kC, s0, 0);
  v16h QA1 = load_A_f16(qbase, kC, s0, 32);

  // ---------------- pass 1: online softmax statistics ----------------
  float m_[8], Z_[8];
#pragma unroll
  for (int i = 0; i < 8; ++i) { m_[i] = -3.0e38f; Z_[i] = 0.f; }

  for (int lt = 0; lt < kLTiles; ++lt) {
    int l0 = lt * 32;
    float smix[2][8];
    compute_smix(QA0, QA1, kbase, wpre, wslope, pm, sh_e, j, s0, l0, smix);
#pragma unroll
    for (int i = 0; i < 8; ++i) {
      float t = fmaxf(smix[0][i], smix[1][i]);
      t = rowmax16(t);
      float mo = m_[i];
      float mn = fmaxf(mo, t);
      float ps = __expf(smix[0][i] - mn) + __expf(smix[1][i] - mn);
      ps = rowsum16(ps);
      Z_[i] = Z_[i] * __expf(mo - mn) + ps;
      m_[i] = mn;
    }
  }
  float rinvZ[8];
#pragma unroll
  for (int i = 0; i < 8; ++i) rinvZ[i] = 1.0f / Z_[i];

  // ---------------- pass 2: normalized P, W_post mix, P·V ----------------
  v8f accv[4] = {{}, {}, {}, {}};
  for (int lt = 0; lt < kLTiles; ++lt) {
    int l0 = lt * 32;
    float smix[2][8];
    compute_smix(QA0, QA1, kbase, wpre, wslope, pm, sh_e, j, s0, l0, smix);

    float p2[2][8];
#pragma unroll
    for (int c = 0; c < 2; ++c)
#pragma unroll
      for (int i = 0; i < 8; ++i)
        p2[c][i] = __expf(smix[c][i] - m_[i]) * rinvZ[i];

    __syncthreads();
#pragma unroll
    for (int c = 0; c < 2; ++c)
#pragma unroll
      for (int i = 0; i < 8; ++i)
        sh_e[j][hf * 8 + i][c * 16 + cn] = p2[c][i];
    __syncthreads();

#pragma unroll
    for (int c = 0; c < 2; ++c) {
#pragma unroll
      for (int i = 0; i < 8; ++i) {
        float v = 0.f;
#pragma unroll
        for (int hh = 0; hh < 8; ++hh)
          v = fmaf(wpost[hh], sh_e[hh][hf * 8 + i][c * 16 + cn], v);
        sh_a[j][hf * 8 + i][c * 16 + cn] = (half_t)v;
      }
    }
    __syncthreads();

    v16h PA = load_A_f16(&sh_a[j][0][0], 32, 0, 0);
    __builtin_prefetch(vbase + (size_t)cn * kSPAD + l0 + 64, 0, 1);
#pragma unroll
    for (int n = 0; n < 4; ++n)
      accv[n] = wmma_ff(PA, load_B16(vbase + (size_t)(n * 16) * kSPAD + l0,
                                     kSPAD), accv[n]);
  }

#pragma unroll
  for (int n = 0; n < 4; ++n)
    store_C_f16(out16 + (size_t)b * kS * kC, kC, s0, j * kD + n * 16, accv[n]);
}

// ---------------------------------------------------------------------------
// K3: att = out @ W_fc^T + b_fc;  h = LN(x + att) * g + b;  h16 [b,s,c]
//   A panel (16x512, contiguous) staged into LDS once via TDM.
// ---------------------------------------------------------------------------
__global__ __launch_bounds__(256) void fc_ln_kernel(
    const half_t* __restrict__ out16, const half_t* __restrict__ wfc16,
    const float* __restrict__ b_fc, const float* __restrict__ x,
    const float* __restrict__ ln_g, const float* __restrict__ ln_b,
    half_t* __restrict__ h16) {
  __shared__ __align__(16) half_t sh_pan[16 * kC];   // 16 KB A panel
  __shared__ float ssum[8][16];
  __shared__ float ssq[8][16];

  int wave = threadIdx.x >> 5;
  int lane = threadIdx.x & 31;
  int cn = lane & 15, hf = lane >> 4;
  int qt = blockIdx.x & 63;
  int b = blockIdx.x >> 6;
  int s0 = qt * 16;
  int o = wave;                             // 64-column chunk per wave

  stage_to_lds(sh_pan, out16 + ((size_t)b * kS + s0) * kC, 16 * kC);
  __syncthreads();

  v8f acc[4] = {{}, {}, {}, {}};
  for (int k0 = 0; k0 < kC; k0 += 32) {
    v16h A = load_A_f16(sh_pan, kC, 0, k0);
    __builtin_prefetch(wfc16 + (size_t)(o * 64) * kC + k0 + 256, 0, 1);
#pragma unroll
    for (int n = 0; n < 4; ++n)
      acc[n] = wmma_ff(A, load_B16(wfc16 + (size_t)(o * 64 + n * 16) * kC + k0,
                                   kC), acc[n]);
  }

  float vals[4][8];
  float prow[8], prow2[8];
#pragma unroll
  for (int i = 0; i < 8; ++i) { prow[i] = 0.f; prow2[i] = 0.f; }
#pragma unroll
  for (int n = 0; n < 4; ++n) {
#pragma unroll
    for (int i = 0; i < 8; ++i) {
      int row = s0 + hf * 8 + i;
      int col = o * 64 + n * 16 + cn;
      float v = acc[n][i] + b_fc[col] + x[((size_t)b * kS + row) * kC + col];
      vals[n][i] = v;
      prow[i] += v;
      prow2[i] += v * v;
    }
  }
#pragma unroll
  for (int i = 0; i < 8; ++i) { prow[i] = rowsum16(prow[i]); prow2[i] = rowsum16(prow2[i]); }
  if ((lane & 15) == 0) {
#pragma unroll
    for (int i = 0; i < 8; ++i) {
      ssum[o][hf * 8 + i] = prow[i];
      ssq[o][hf * 8 + i] = prow2[i];
    }
  }
  __syncthreads();

  float mu[8], rstd[8];
#pragma unroll
  for (int i = 0; i < 8; ++i) {
    int ri = hf * 8 + i;
    float s = 0.f, s2 = 0.f;
#pragma unroll
    for (int w = 0; w < 8; ++w) { s += ssum[w][ri]; s2 += ssq[w][ri]; }
    float mm = s * (1.0f / kC);
    float var = s2 * (1.0f / kC) - mm * mm;
    mu[i] = mm;
    rstd[i] = rsqrtf(var + 1e-5f);
  }
#pragma unroll
  for (int n = 0; n < 4; ++n) {
#pragma unroll
    for (int i = 0; i < 8; ++i) {
      int row = s0 + hf * 8 + i;
      int col = o * 64 + n * 16 + cn;
      float v = (vals[n][i] - mu[i]) * rstd[i] * ln_g[col] + ln_b[col];
      h16[((size_t)b * kS + row) * kC + col] = (half_t)v;
    }
  }
}

// ---------------------------------------------------------------------------
// K4/K5: causal conv (K=3) as 3 shifted WMMA GEMMs + masked ReLU epilogues.
//   A panel = rows [s0-2, s0+16) staged into LDS via TDM (first 2 rows
//   zero-filled at the sequence start -> causal zero padding for free).
//   is_last==0: write conv1 result (f16). is_last==1: add residual, write f32.
// ---------------------------------------------------------------------------
__global__ __launch_bounds__(256) void conv_kernel(
    const half_t* __restrict__ in16, const half_t* __restrict__ wB,
    const float* __restrict__ bias, const unsigned char* __restrict__ pad_mask,
    const half_t* __restrict__ res16, half_t* __restrict__ o16,
    float* __restrict__ ofinal, int is_last) {
  __shared__ __align__(16) half_t sh_pan[(kK - 1 + 16) * kC];   // 18 KB

  int wave = threadIdx.x >> 5;
  int lane = threadIdx.x & 31;
  int cn = lane & 15, hf = lane >> 4;
  int qt = blockIdx.x & 63;
  int b = blockIdx.x >> 6;
  int s0 = qt * 16;
  int o = wave;

  int lead = (s0 == 0) ? (kK - 1) : 0;       // rows to zero-fill (causal pad)
  if (lead) {
    for (int i = threadIdx.x; i < lead * kC; i += blockDim.x)
      sh_pan[i] = (half_t)0.f;
  }
  const half_t* gsrc = in16 + ((size_t)b * kS + (s0 - (kK - 1) + lead)) * kC;
  stage_to_lds(sh_pan + lead * kC, gsrc, (unsigned)((kK - 1 + 16 - lead) * kC));
  __syncthreads();

  v8f acc[4] = {{}, {}, {}, {}};
#pragma unroll
  for (int t = 0; t < kK; ++t) {
    const half_t* Wt = wB + (size_t)t * kC * kC;
    for (int k0 = 0; k0 < kC; k0 += 32) {
      v16h A = load_A_f16(sh_pan, kC, t, k0);   // panel row t == s0-2+t
      __builtin_prefetch(Wt + (size_t)(o * 64) * kC + k0 + 256, 0, 1);
#pragma unroll
      for (int n = 0; n < 4; ++n)
        acc[n] = wmma_ff(A, load_B16(Wt + (size_t)(o * 64 + n * 16) * kC + k0,
                                     kC), acc[n]);
    }
  }

#pragma unroll
  for (int n = 0; n < 4; ++n) {
#pragma unroll
    for (int i = 0; i < 8; ++i) {
      int row = s0 + hf * 8 + i;
      int col = o * 64 + n * 16 + cn;
      bool msk = pad_mask[(size_t)b * kS + row] != 0;
      float v = acc[n][i] + bias[col];
      v = msk ? 0.f : v;                 // where(m, 0, conv+b)
      v = fmaxf(v, 0.f);                 // relu
      if (!is_last) {
        o16[((size_t)b * kS + row) * kC + col] = (half_t)v;
      } else {
        float rres = msk ? 0.f : (float)res16[((size_t)b * kS + row) * kC + col];
        float oo = v + rres;             // o + res
        oo = fmaxf(oo, 0.f);             // relu
        oo = msk ? 0.f : oo;             // where(m, 0, .)
        ofinal[((size_t)b * kS + row) * kC + col] = oo;
      }
    }
  }
}

// ---------------------------------------------------------------------------
// Host orchestration
// ---------------------------------------------------------------------------
extern "C" void kernel_launch(void* const* d_in, const int* in_sizes, int n_in,
                              void* d_out, int out_size, void* d_ws,
                              size_t ws_size, hipStream_t stream) {
  const float* x = (const float*)d_in[0];
  const unsigned char* pad_mask = (const unsigned char*)d_in[1];
  const float* Wq = (const float*)d_in[2];
  const float* Wk = (const float*)d_in[3];
  const float* Wv = (const float*)d_in[4];
  const float* W_fc = (const float*)d_in[5];
  const float* b_fc = (const float*)d_in[6];
  const float* W_pre = (const float*)d_in[7];
  const float* W_post = (const float*)d_in[8];
  const float* p_keys = (const float*)d_in[9];
  const float* p_values = (const float*)d_in[10];
  const float* ln_g = (const float*)d_in[11];
  const float* ln_b = (const float*)d_in[12];
  const float* conv1_w = (const float*)d_in[13];
  const float* conv1_b = (const float*)d_in[14];
  const float* conv2_w = (const float*)d_in[15];
  const float* conv2_b = (const float*)d_in[16];
  float* out = (float*)d_out;

  char* ws = (char*)d_ws;
  size_t off = 0;
  auto take = [&](size_t bytes) -> void* {
    void* p = ws + off;
    off = (off + bytes + 255) & ~(size_t)255;
    return p;
  };
  half_t* q16 = (half_t*)take((size_t)kB * kS * kC * 2);
  half_t* k16 = (half_t*)take((size_t)kB * kSPAD * kC * 2);
  half_t* vT16 = (half_t*)take((size_t)kB * kH * kD * kSPAD * 2);
  half_t* o16a = (half_t*)take((size_t)kB * kS * kC * 2);   // attn out
  half_t* h16 = (half_t*)take((size_t)kB * kS * kC * 2);    // post-LN
  half_t* c116 = (half_t*)take((size_t)kB * kS * kC * 2);   // conv1 out
  half_t* wq16 = (half_t*)take((size_t)kD * kD * 2);
  half_t* wk16 = (half_t*)take((size_t)kD * kD * 2);
  half_t* wv16 = (half_t*)take((size_t)kD * kD * 2);
  half_t* wfc16 = (half_t*)take((size_t)kC * kC * 2);
  half_t* wb1 = (half_t*)take((size_t)kK * kC * kC * 2);
  half_t* wb2 = (half_t*)take((size_t)kK * kC * kC * 2);

  // --- weight prep ---
  convert_f16_kernel<<<(kD * kD + 255) / 256, 256, 0, stream>>>(Wq, wq16, kD * kD);
  convert_f16_kernel<<<(kD * kD + 255) / 256, 256, 0, stream>>>(Wk, wk16, kD * kD);
  convert_f16_kernel<<<(kD * kD + 255) / 256, 256, 0, stream>>>(Wv, wv16, kD * kD);
  convert_f16_kernel<<<(kC * kC + 255) / 256, 256, 0, stream>>>(W_fc, wfc16, kC * kC);
  pack_conv_kernel<<<(kK * kC * kC + 255) / 256, 256, 0, stream>>>(conv1_w, wb1);
  pack_conv_kernel<<<(kK * kC * kC + 255) / 256, 256, 0, stream>>>(conv2_w, wb2);

  // --- persistent K/V + padding ---
  persist_kv_kernel<<<1024, 256, 0, stream>>>(p_keys, p_values, k16, vT16);

  // --- pipeline ---
  qkv_kernel<<<512, 256, 0, stream>>>(x, wq16, wk16, wv16, q16, k16, vT16);
  attn_kernel<<<kB * (kS / 16), 256, 0, stream>>>(q16, k16, vT16, W_pre, W_post,
                                                  pad_mask, o16a);
  fc_ln_kernel<<<kB * (kS / 16), 256, 0, stream>>>(o16a, wfc16, b_fc, x, ln_g,
                                                   ln_b, h16);
  conv_kernel<<<kB * (kS / 16), 256, 0, stream>>>(h16, wb1, conv1_b, pad_mask,
                                                  nullptr, c116, nullptr, 0);
  conv_kernel<<<kB * (kS / 16), 256, 0, stream>>>(c116, wb2, conv2_b, pad_mask,
                                                  h16, nullptr, out, 1);
}
